// BiLSTM_CRF_60627758350850
// MI455X (gfx1250) — matrix operand order repaired and checked by reading, not compiled, not run on hardware
//
#include <hip/hip_runtime.h>
#include <hip/hip_bf16.h>
#include <math.h>
#include <stdint.h>

typedef __attribute__((ext_vector_type(16))) _Float16 v16h;
typedef __attribute__((ext_vector_type(8)))  _Float16 v8h;
typedef __attribute__((ext_vector_type(8)))  float    v8f;
typedef uint32_t v4u __attribute__((ext_vector_type(4)));
typedef int      v8i __attribute__((ext_vector_type(8)));
typedef int      v4i __attribute__((ext_vector_type(4)));

#define S_LEN 4096
#define D_EMB 1024
#define H_HID 512
#define G4H   2048   // 4*H
#define T_TAG 20
#define NPAD  32

// GEMM blocking
#define BM 64
#define BN 256
#define BK 32

// ---------------------------------------------------------------------------
// Prep kernels
// ---------------------------------------------------------------------------
__global__ void k_embed_cast(const int* __restrict__ sent,
                             const float* __restrict__ embed,
                             _Float16* __restrict__ emb16) {
  int i = blockIdx.x * blockDim.x + threadIdx.x;     // over S*D (exact)
  int s = i >> 10;
  int d = i & 1023;
  emb16[i] = (_Float16)embed[(size_t)sent[s] * D_EMB + d];
}

__global__ void k_cast_f16(const float* __restrict__ in,
                           _Float16* __restrict__ out, int n) {
  int i = blockIdx.x * blockDim.x + threadIdx.x;
  if (i < n) out[i] = (_Float16)in[i];
}

__global__ void k_tag_prep(const float* __restrict__ Wtag,
                           const float* __restrict__ btag,
                           _Float16* __restrict__ Wtag16,
                           float* __restrict__ btag_pad) {
  int i = blockIdx.x * blockDim.x + threadIdx.x;      // over NPAD*1024 (exact)
  int r = i >> 10;
  int c = i & 1023;
  Wtag16[i] = (r < T_TAG) ? (_Float16)Wtag[r * (2 * H_HID) + c] : (_Float16)0.f;
  if (i < NPAD) btag_pad[i] = (i < T_TAG) ? btag[i] : 0.f;
}

__global__ void k_transpose(const float* __restrict__ in,   // [G4H, H]
                            float* __restrict__ out) {      // [H, G4H]
  int i = blockIdx.x * blockDim.x + threadIdx.x;      // over G4H*H (exact)
  int r = i >> 9;
  int k = i & 511;
  out[(size_t)k * G4H + r] = in[i];
}

// ---------------------------------------------------------------------------
// TDM 2D tile load: global (row-major, stride in elements) -> LDS contiguous.
// Descriptor packing per CDNA5 ISA sec 8.3/8.4 (D# group0/group1), 2D tensor:
// group2/group3 zero. data_size=1 (2-byte elements).
// ---------------------------------------------------------------------------
__device__ __forceinline__ void tdm_load_2d(const _Float16* gptr,
                                            uint32_t stride_elems,
                                            uint32_t rows, uint32_t cols,
                                            uint32_t lds_byte_addr) {
  uint64_t ga = (uint64_t)(size_t)gptr;
  v4u g0;
  g0[0] = 1u;                                   // count=1, user descriptor
  g0[1] = lds_byte_addr;                        // lds_addr (bytes)
  g0[2] = (uint32_t)ga;                         // global_addr[31:0]
  g0[3] = (uint32_t)((ga >> 32) & 0x01FFFFFFu) | (2u << 30);  // addr[56:32] | type=2
  const uint32_t TD = 1u << 20;                 // large tensor dims (no OOB clipping)
  v8i g1;
  g1[0] = (int)(1u << 16);                      // data_size=1 (2B); wg_mask=0; no pad/iter
  g1[1] = (int)((TD & 0xFFFFu) << 16);          // [31:16]=tensor_dim0[15:0]
  g1[2] = (int)(((TD >> 16) & 0xFFFFu) | ((TD & 0xFFFFu) << 16)); // td0[31:16] | td1[15:0]
  g1[3] = (int)(((TD >> 16) & 0xFFFFu) | (cols << 16));           // td1[31:16] | tile_dim0
  g1[4] = (int)(rows & 0xFFFFu);                // tile_dim1; tile_dim2=0
  g1[5] = (int)stride_elems;                    // tensor_dim0_stride[31:0]
  g1[6] = 0;                                    // stride[47:32] | tensor_dim1_stride lo
  g1[7] = 0;
  v4i gz = {};
#if defined(__clang_major__) && (__clang_major__ >= 23)
  v8i gz8 = {};
  __builtin_amdgcn_tensor_load_to_lds(g0, g1, gz, gz, gz8, 0);
#else
  __builtin_amdgcn_tensor_load_to_lds(g0, g1, gz, gz, 0);
#endif
}

// ---------------------------------------------------------------------------
// TDM + LDS double-buffered WMMA GEMM: C[M,N] = A[M,K] * B[N,K]^T + bias[N]
// WG = 256 threads (8 waves). WG tile 64x256, wave tile 32x64 (2x4 WMMA tiles).
// Each wave TDM-loads its A slice (8 rows) and B slice (32 rows) per K-step.
// ---------------------------------------------------------------------------
__global__ __launch_bounds__(256)
void k_gemm_tdm_wmma(const _Float16* __restrict__ A, int lda,
                     const _Float16* __restrict__ B, int ldb,
                     const float* __restrict__ bias,
                     float* __restrict__ C, int ldc,
                     int tilesN, int K) {
  __shared__ _Float16 Abuf[2][BM * BK];   // 2 x 4 KB
  __shared__ _Float16 Bbuf[2][BN * BK];   // 2 x 16 KB
  const int bm = blockIdx.x / tilesN;
  const int bn = blockIdx.x % tilesN;
  const int m0 = bm * BM;
  const int n0 = bn * BN;
  const int tid  = threadIdx.x;
  const int lane = tid & 31;
  const int w    = tid >> 5;        // wave 0..7
  const int wm   = w & 1;           // wave M index: rows wm*32
  const int wn   = w >> 1;          // wave N index: cols wn*64
  const int l16  = lane & 15;
  const int hi   = lane >> 4;

  const _Float16* Ag = A + (size_t)(m0 + w * 8) * lda;    // wave's 8-row A slice
  const _Float16* Bg = B + (size_t)(n0 + w * 32) * ldb;   // wave's 32-row B slice

  const int steps = K / BK;
  // prologue: fill buffer 0
  tdm_load_2d(Ag, (uint32_t)lda, 8,  BK, (uint32_t)(size_t)&Abuf[0][w * 8 * BK]);
  tdm_load_2d(Bg, (uint32_t)ldb, 32, BK, (uint32_t)(size_t)&Bbuf[0][w * 32 * BK]);

  v8f acc[2][4];
#pragma unroll
  for (int i = 0; i < 2; ++i)
#pragma unroll
    for (int j = 0; j < 4; ++j) acc[i][j] = (v8f){};

  for (int s = 0; s < steps; ++s) {
    if (s + 1 < steps) {
      const int p = (s + 1) & 1;
      const int kk = (s + 1) * BK;
      tdm_load_2d(Ag + kk, (uint32_t)lda, 8,  BK,
                  (uint32_t)(size_t)&Abuf[p][w * 8 * BK]);
      tdm_load_2d(Bg + kk, (uint32_t)ldb, 32, BK,
                  (uint32_t)(size_t)&Bbuf[p][w * 32 * BK]);
      // per-wave in-order TDM: <=2 outstanding means this step's 2 loads done
      __builtin_amdgcn_s_wait_tensorcnt(2);
    } else {
      __builtin_amdgcn_s_wait_tensorcnt(0);
    }
    __syncthreads();   // all waves' slices resident

    const _Float16* Ab = Abuf[s & 1];
    const _Float16* Bb = Bbuf[s & 1];
    v16h afrag[2], bfrag[4];
#pragma unroll
    for (int i = 0; i < 2; ++i) {
      const _Float16* ar = Ab + (wm * 32 + i * 16 + l16) * BK;
      v8h lo = *(const v8h*)(ar + hi * 8);
      v8h hh = *(const v8h*)(ar + 16 + hi * 8);
      afrag[i] = __builtin_shufflevector(lo, hh,
                   0,1,2,3,4,5,6,7,8,9,10,11,12,13,14,15);
    }
#pragma unroll
    for (int j = 0; j < 4; ++j) {
      const _Float16* br = Bb + (wn * 64 + j * 16 + l16) * BK;
      v8h lo = *(const v8h*)(br + hi * 8);
      v8h hh = *(const v8h*)(br + 16 + hi * 8);
      bfrag[j] = __builtin_shufflevector(lo, hh,
                   0,1,2,3,4,5,6,7,8,9,10,11,12,13,14,15);
    }
#pragma unroll
    for (int i = 0; i < 2; ++i)
#pragma unroll
      for (int j = 0; j < 4; ++j)
        acc[i][j] = __builtin_amdgcn_wmma_f32_16x16x32_f16(
                        false, afrag[i], false, bfrag[j], (short)0,
                        acc[i][j], false, false);
    __syncthreads();   // protect buffers before next step's TDM overwrite
  }

#pragma unroll
  for (int i = 0; i < 2; ++i)
#pragma unroll
    for (int j = 0; j < 4; ++j) {
      const int col = n0 + wn * 64 + j * 16 + l16;
      const float bv = bias[col];
      float* crow = C + (size_t)(m0 + wm * 32 + i * 16 + hi * 8) * ldc + col;
#pragma unroll
      for (int e = 0; e < 8; ++e) crow[(size_t)e * ldc] = acc[i][j][e] + bv;
    }
}

// ---------------------------------------------------------------------------
// Direct WMMA GEMM for the small tag projection (N=32): one 16x16 tile / wave.
// ---------------------------------------------------------------------------
__global__ __launch_bounds__(256)
void k_gemm_f16_wmma(const _Float16* __restrict__ A, int lda,
                     const _Float16* __restrict__ B, int ldb,
                     const float* __restrict__ bias,
                     float* __restrict__ C, int ldc,
                     int tiles_n, int K) {
  const int lane = threadIdx.x & 31;
  const int wave = threadIdx.x >> 5;
  const int tile = blockIdx.x * 8 + wave;
  const int m0 = (tile / tiles_n) * 16;
  const int n0 = (tile % tiles_n) * 16;
  const int l16 = lane & 15;
  const int hi  = lane >> 4;
  const _Float16* arow = A + (size_t)(m0 + l16) * lda;
  const _Float16* brow = B + (size_t)(n0 + l16) * ldb;
  const int akoff = hi * 8;
  const int bkoff = hi * 8;

  v8f c = {};
  for (int kk = 0; kk < K; kk += 32) {
    v8h a_lo = *(const v8h*)(arow + kk + akoff);
    v8h a_hi = *(const v8h*)(arow + kk + akoff + 16);
    v8h b_lo = *(const v8h*)(brow + kk + bkoff);
    v8h b_hi = *(const v8h*)(brow + kk + bkoff + 16);
    v16h a = __builtin_shufflevector(a_lo, a_hi,
               0,1,2,3,4,5,6,7,8,9,10,11,12,13,14,15);
    v16h b = __builtin_shufflevector(b_lo, b_hi,
               0,1,2,3,4,5,6,7,8,9,10,11,12,13,14,15);
    c = __builtin_amdgcn_wmma_f32_16x16x32_f16(
            false, a, false, b, (short)0, c, false, false);
  }
  const float bv = bias[n0 + l16];
  float* crow = C + (size_t)(m0 + hi * 8) * ldc + n0 + l16;
#pragma unroll
  for (int e = 0; e < 8; ++e) crow[(size_t)e * ldc] = c[e] + bv;
}

// ---------------------------------------------------------------------------
// Sequential LSTM recurrence (1 WG per direction, 1024 threads).
// ---------------------------------------------------------------------------
__device__ __forceinline__ float sigmf(float x) { return 1.f / (1.f + expf(-x)); }

__global__ __launch_bounds__(1024)
void k_lstm(const float* __restrict__ Gf, const float* __restrict__ Gb,
            const float* __restrict__ WhTf, const float* __restrict__ WhTb,
            const float* __restrict__ h0, const float* __restrict__ c0,
            _Float16* __restrict__ hcat) {
  __shared__ float h_lds[H_HID];
  __shared__ float g_lds[G4H];
  const int dir = blockIdx.x;
  const float* __restrict__ G   = dir ? Gb   : Gf;
  const float* __restrict__ WhT = dir ? WhTb : WhTf;
  const int tid = threadIdx.x;

  float c = 0.f;
  if (tid < H_HID) {
    h_lds[tid] = h0[dir * H_HID + tid];
    c          = c0[dir * H_HID + tid];
  }
  __syncthreads();

  const int row0 = tid;
  const int row1 = tid + 1024;
  for (int t = 0; t < S_LEN; ++t) {
    const int trow = dir ? (S_LEN - 1 - t) : t;
    const float* g0 = G + (size_t)trow * G4H;
    float acc0 = g0[row0];
    float acc1 = g0[row1];
#pragma unroll 8
    for (int k = 0; k < H_HID; ++k) {
      const float hk = h_lds[k];
      const float* wp = WhT + (size_t)k * G4H;
      acc0 = fmaf(wp[row0], hk, acc0);
      acc1 = fmaf(wp[row1], hk, acc1);
    }
    g_lds[row0] = acc0;
    g_lds[row1] = acc1;
    __syncthreads();
    if (tid < H_HID) {
      const float gi = g_lds[tid];
      const float gf = g_lds[H_HID + tid];
      const float gc = g_lds[2 * H_HID + tid];
      const float go = g_lds[3 * H_HID + tid];
      c = sigmf(gf) * c + sigmf(gi) * tanhf(gc);
      const float h = sigmf(go) * tanhf(c);
      h_lds[tid] = h;
      hcat[(size_t)trow * (2 * H_HID) + dir * H_HID + tid] = (_Float16)h;
    }
    __syncthreads();
  }
}

// ---------------------------------------------------------------------------
// Viterbi: one wave32, lane = tag.
// ---------------------------------------------------------------------------
__global__ void k_viterbi(const float* __restrict__ feats,   // [S, NPAD]
                          const float* __restrict__ trans,   // [T, T]
                          const int* __restrict__ start_idx,
                          const int* __restrict__ stop_idx,
                          int* __restrict__ bptrs,           // [S, NPAD]
                          float* __restrict__ out) {
  const float NEG = -10000.f;
  const int lane = threadIdx.x;
  const int start = *start_idx;
  const int stop  = *stop_idx;

  float tr[T_TAG];
#pragma unroll
  for (int p = 0; p < T_TAG; ++p)
    tr[p] = (lane < T_TAG) ? trans[lane * T_TAG + p] : NEG;

  float fv = (lane == start) ? 0.f : NEG;
  for (int t = 0; t < S_LEN; ++t) {
    float best = -3.4e38f;
    int bp = 0;
#pragma unroll
    for (int p = 0; p < T_TAG; ++p) {
      const float v = __shfl(fv, p, 32) + tr[p];
      if (v > best) { best = v; bp = p; }
    }
    bptrs[t * NPAD + lane] = bp;
    fv = best + feats[t * NPAD + lane];
  }

  float term = (lane < T_TAG) ? fv + trans[stop * T_TAG + lane] : -3.4e38f;
  int idx = lane;
#pragma unroll
  for (int off = 16; off > 0; off >>= 1) {
    const float ov = __shfl_xor(term, off, 32);
    const int   oi = __shfl_xor(idx,  off, 32);
    if (ov > term || (ov == term && oi < idx)) { term = ov; idx = oi; }
  }

  if (lane == 0) {
    out[0] = term;
    int cur = idx;
    out[1 + (S_LEN - 1)] = (float)cur;
    for (int t = S_LEN - 1; t >= 1; --t) {
      cur = bptrs[t * NPAD + cur];
      out[1 + (t - 1)] = (float)cur;
    }
  }
}

// ---------------------------------------------------------------------------
// Host orchestration
// ---------------------------------------------------------------------------
extern "C" void kernel_launch(void* const* d_in, const int* in_sizes, int n_in,
                              void* d_out, int out_size, void* d_ws, size_t ws_size,
                              hipStream_t stream) {
  const int*   sentence = (const int*)  d_in[0];
  const float* embed    = (const float*)d_in[1];
  const float* W_ih_f   = (const float*)d_in[2];
  const float* W_hh_f   = (const float*)d_in[3];
  const float* b_f      = (const float*)d_in[4];
  const float* W_ih_b   = (const float*)d_in[5];
  const float* W_hh_b   = (const float*)d_in[6];
  const float* b_b      = (const float*)d_in[7];
  const float* h0       = (const float*)d_in[8];
  const float* c0       = (const float*)d_in[9];
  const float* W_tag    = (const float*)d_in[10];
  const float* b_tag    = (const float*)d_in[11];
  const float* trans    = (const float*)d_in[12];
  const int*   start_i  = (const int*)  d_in[13];
  const int*   stop_i   = (const int*)  d_in[14];
  float* out = (float*)d_out;

  char* ws = (char*)d_ws;
  size_t off = 0;
  auto alloc = [&](size_t bytes) -> char* {
    char* p = ws + off;
    off = (off + bytes + 255) & ~(size_t)255;
    return p;
  };
  _Float16* emb16   = (_Float16*)alloc((size_t)S_LEN * D_EMB * 2);
  _Float16* wf16    = (_Float16*)alloc((size_t)G4H * D_EMB * 2);
  _Float16* wb16    = (_Float16*)alloc((size_t)G4H * D_EMB * 2);
  _Float16* wtag16  = (_Float16*)alloc((size_t)NPAD * (2 * H_HID) * 2);
  float*    btagp   = (float*)   alloc((size_t)NPAD * 4);
  float*    Gf      = (float*)   alloc((size_t)S_LEN * G4H * 4);
  float*    Gb      = (float*)   alloc((size_t)S_LEN * G4H * 4);
  float*    WhTf    = (float*)   alloc((size_t)H_HID * G4H * 4);
  float*    WhTb    = (float*)   alloc((size_t)H_HID * G4H * 4);
  _Float16* hcat    = (_Float16*)alloc((size_t)S_LEN * (2 * H_HID) * 2);
  float*    feats   = (float*)   alloc((size_t)S_LEN * NPAD * 4);
  int*      bptrs   = (int*)     alloc((size_t)S_LEN * NPAD * 4);
  (void)ws_size; (void)in_sizes; (void)n_in; (void)out_size;

  // 1) prep
  k_embed_cast<<<(S_LEN * D_EMB) / 256, 256, 0, stream>>>(sentence, embed, emb16);
  k_cast_f16<<<(G4H * D_EMB + 255) / 256, 256, 0, stream>>>(W_ih_f, wf16, G4H * D_EMB);
  k_cast_f16<<<(G4H * D_EMB + 255) / 256, 256, 0, stream>>>(W_ih_b, wb16, G4H * D_EMB);
  k_tag_prep<<<(NPAD * (2 * H_HID)) / 256, 256, 0, stream>>>(W_tag, b_tag, wtag16, btagp);
  k_transpose<<<(G4H * H_HID) / 256, 256, 0, stream>>>(W_hh_f, WhTf);
  k_transpose<<<(G4H * H_HID) / 256, 256, 0, stream>>>(W_hh_b, WhTb);

  // 2) input projections: TDM->LDS double-buffered WMMA GEMMs
  {
    const int tilesN = G4H / BN;                       // 8
    const int blocks = (S_LEN / BM) * tilesN;          // 512
    k_gemm_tdm_wmma<<<blocks, 256, 0, stream>>>(emb16, D_EMB, wf16, D_EMB,
                                                b_f, Gf, G4H, tilesN, D_EMB);
    k_gemm_tdm_wmma<<<blocks, 256, 0, stream>>>(emb16, D_EMB, wb16, D_EMB,
                                                b_b, Gb, G4H, tilesN, D_EMB);
  }

  // 3) sequential bidirectional LSTM
  k_lstm<<<2, 1024, 0, stream>>>(Gf, Gb, WhTf, WhTb, h0, c0, hcat);

  // 4) tag projection (small N): direct WMMA GEMM
  {
    const int tiles_n = NPAD / 16;                     // 2
    const int blocks  = (S_LEN / 16) * tiles_n / 8;    // 64
    k_gemm_f16_wmma<<<blocks, 256, 0, stream>>>(hcat, 2 * H_HID, wtag16, 2 * H_HID,
                                                btagp, feats, NPAD, tiles_n, 2 * H_HID);
  }

  // 5) Viterbi decode
  k_viterbi<<<1, 32, 0, stream>>>(feats, trans, start_i, stop_i, bptrs, out);
}
